// QKVAttention_71107478552917
// MI455X (gfx1250) — matrix operand order repaired and checked by reading, not compiled, not run on hardware
//
#include <hip/hip_runtime.h>

typedef __attribute__((ext_vector_type(16))) __bf16 v16bf;
typedef __attribute__((ext_vector_type(8)))  float  v8f;
typedef unsigned short u16;
typedef unsigned int   u32;

// round-to-nearest-even f32 -> bf16 (bit pattern as u16)
__device__ __forceinline__ u16 f2bf(float f) {
    u32 u = __builtin_bit_cast(u32, f);
    u32 r = u + 0x7fffu + ((u >> 16) & 1u);
    return (u16)(r >> 16);
}
__device__ __forceinline__ float bf2f(u16 h) {
    return __builtin_bit_cast(float, (u32)h << 16);
}

union BfOp { int4 q[2]; v16bf v; };

#define WMMA_BF16(A, B, C) \
    __builtin_amdgcn_wmma_f32_16x16x32_bf16(false, (A), false, (B), (short)0, (C), false, false)

#define NB   16
#define NN   200
#define NP   208     // padded N (13 * 16)
#define DD   2048
#define HH   2048

// ---------------------------------------------------------------------------
// Pre-pass A: convert activations f32 -> bf16, pad rows [200,208) with zeros.
//   blockIdx.z selects {query,key,value,loc}.
// ---------------------------------------------------------------------------
__global__ __launch_bounds__(256) void kc_x(
    const float* __restrict__ qin, const float* __restrict__ kin,
    const float* __restrict__ vin, const float* __restrict__ locin,
    u16* __restrict__ Xq, u16* __restrict__ Xk,
    u16* __restrict__ Xv, u16* __restrict__ Xl)
{
    const float* src; u16* dst;
    switch (blockIdx.z) {
        case 0:  src = qin;   dst = Xq; break;
        case 1:  src = kin;   dst = Xk; break;
        case 2:  src = vin;   dst = Xv; break;
        default: src = locin; dst = Xl; break;
    }
    size_t i4 = ((size_t)blockIdx.x * 256 + threadIdx.x) * 4; // < 16*208*2048
    int b   = (int)(i4 / ((size_t)NP * HH));
    int rem = (int)(i4 % ((size_t)NP * HH));
    int n = rem >> 11;
    int h = rem & (HH - 1);
    ushort4 o = make_ushort4(0, 0, 0, 0);
    if (n < NN) {
        float4 xv = *(const float4*)(src + ((size_t)(b * NN + n)) * DD + h);
        o.x = f2bf(xv.x); o.y = f2bf(xv.y); o.z = f2bf(xv.z); o.w = f2bf(xv.w);
    }
    *(ushort4*)(dst + i4) = o;
}

// ---------------------------------------------------------------------------
// Pre-pass B: weights f32 [d][h] -> bf16 transposed WT[h][d].
// 64x64 tile staged in LDS via async global->LDS b128 copies (ASYNCcnt),
// then converted + transposed out with coalesced b128 stores.
//   blockIdx.z selects {Wq,Wk,Wv,Wloc,Wlk}.
// ---------------------------------------------------------------------------
__global__ __launch_bounds__(256) void kc_w(
    const float* __restrict__ Wq,  const float* __restrict__ Wk,
    const float* __restrict__ Wv,  const float* __restrict__ Wloc,
    const float* __restrict__ Wlk,
    u16* __restrict__ WTq, u16* __restrict__ WTk, u16* __restrict__ WTv,
    u16* __restrict__ WTloc, u16* __restrict__ WTlk)
{
    const float* W; u16* WT;
    switch (blockIdx.z) {
        case 0:  W = Wq;   WT = WTq;   break;
        case 1:  W = Wk;   WT = WTk;   break;
        case 2:  W = Wv;   WT = WTv;   break;
        case 3:  W = Wloc; WT = WTloc; break;
        default: W = Wlk;  WT = WTlk;  break;
    }
    const int c0 = blockIdx.x * 64;   // h columns
    const int r0 = blockIdx.y * 64;   // d rows
    const int t  = threadIdx.x;

    __shared__ float tile[64][68];    // row stride 272B (16B-aligned)

#pragma unroll
    for (int i = 0; i < 4; ++i) {
        int idx  = t + 256 * i;       // 0..1023 float4 slots
        int rowl = idx >> 4, cg = idx & 15;
        const float* gp = W + ((size_t)(r0 + rowl)) * HH + c0 + cg * 4;
        u32 ldsoff = (u32)(uintptr_t)&tile[rowl][cg * 4];
        asm volatile("global_load_async_to_lds_b128 %0, %1, off"
                     :: "v"(ldsoff), "v"(gp) : "memory");
    }
    asm volatile("s_wait_asynccnt 0" ::: "memory");
    __syncthreads();

    int hl = t >> 2, dseg = t & 3;    // 4 threads per h-row, 16 d each
    __align__(16) u16 buf[16];
#pragma unroll
    for (int j = 0; j < 16; ++j) buf[j] = f2bf(tile[dseg * 16 + j][hl]);
    u16* dst = WT + ((size_t)(c0 + hl)) * DD + r0 + dseg * 16;
    *(int4*)(dst + 0) = *(const int4*)&buf[0];
    *(int4*)(dst + 8) = *(const int4*)&buf[8];
}

// ---------------------------------------------------------------------------
// Stage 1: five projection GEMMs, all-bf16 operands streamed from L2.
//   z=0: lq   = Xq @ WTq^T + bq      -> bf16 [b][NP][H]
//   z=1: lk   = Xk @ WTk^T + bk      -> bf16 [b][NP][H]
//   z=2: lvT  = (Xv @ WTv^T + bv)^T  -> bf16 [b][H][NP]
//   z=3: lloc = Xl @ WTloc^T + bloc  -> bf16 [b][NP][H]
//   z=4: kloc = Xl @ WTlk^T + blk    -> bf16 [b][NP][H]
// 256 threads = 8 waves (2 row groups x 4 col groups); block tile 64x256;
// wave tile 32x64 (2x4 WMMA tiles). A/B operands: per-lane global b128 in
// contiguous k order (k = 16*kh + s) for both A and B. Epilogue coalesced through LDS.
// ---------------------------------------------------------------------------
__global__ __launch_bounds__(256) void k_proj(
    const u16* __restrict__ Xq, const u16* __restrict__ Xk,
    const u16* __restrict__ Xv, const u16* __restrict__ Xl,
    const u16* __restrict__ WTq, const u16* __restrict__ WTk,
    const u16* __restrict__ WTv, const u16* __restrict__ WTloc,
    const u16* __restrict__ WTlk,
    const float* __restrict__ bq,  const float* __restrict__ bk,
    const float* __restrict__ bv,  const float* __restrict__ bloc,
    const float* __restrict__ blk,
    u16* __restrict__ lq, u16* __restrict__ lko, u16* __restrict__ lvT,
    u16* __restrict__ lloco, u16* __restrict__ kloco)
{
    const int z = blockIdx.z;
    const u16 *X, *WT; const float* bias; u16* outp;
    switch (z) {
        case 0:  X = Xq; WT = WTq;   bias = bq;   outp = lq;    break;
        case 1:  X = Xk; WT = WTk;   bias = bk;   outp = lko;   break;
        case 2:  X = Xv; WT = WTv;   bias = bv;   outp = lvT;   break;
        case 3:  X = Xl; WT = WTloc; bias = bloc; outp = lloco; break;
        default: X = Xl; WT = WTlk;  bias = blk;  outp = kloco; break;
    }
    const int hb    = blockIdx.x * 256;
    const int b     = blockIdx.y >> 2;
    const int nbase = (blockIdx.y & 3) * 64;
    const int tid   = threadIdx.x;
    const int lane  = tid & 31;
    const int w     = tid >> 5;
    const int wr    = w >> 2, wc = w & 3;
    const int row0  = lane & 15, kh = lane >> 4;

    __shared__ u16 Es[64][264];   // epilogue staging (row stride 528B)

    // A row offsets (clamped into padded region; pad rows are zeros)
    int r0i = nbase + 32 * wr +  0 + row0; if (r0i > NP - 1) r0i = NP - 1;
    int r1i = nbase + 32 * wr + 16 + row0; if (r1i > NP - 1) r1i = NP - 1;
    const size_t ao0 = ((size_t)(b * NP) + r0i) * DD;
    const size_t ao1 = ((size_t)(b * NP) + r1i) * DD;
    // B row offsets (h columns of the output)
    int hcol[4];
    size_t bo[4];
#pragma unroll
    for (int ct = 0; ct < 4; ++ct) {
        hcol[ct] = hb + 64 * wc + 16 * ct + row0;
        bo[ct]   = (size_t)hcol[ct] * DD;
    }

    v8f acc[2][4];
#pragma unroll
    for (int rt = 0; rt < 2; ++rt)
#pragma unroll
        for (int ct = 0; ct < 4; ++ct) acc[rt][ct] = (v8f){};

    for (int kb = 0; kb < DD; kb += 32) {
        BfOp A0, A1, Bt[4];
        A0.q[0] = *(const int4*)(X + ao0 + kb + 16 * kh);
        A0.q[1] = *(const int4*)(X + ao0 + kb + 16 * kh + 8);
        A1.q[0] = *(const int4*)(X + ao1 + kb + 16 * kh);
        A1.q[1] = *(const int4*)(X + ao1 + kb + 16 * kh + 8);
#pragma unroll
        for (int ct = 0; ct < 4; ++ct) {
            Bt[ct].q[0] = *(const int4*)(WT + bo[ct] + kb + 16 * kh);
            Bt[ct].q[1] = *(const int4*)(WT + bo[ct] + kb + 16 * kh + 8);
        }
#pragma unroll
        for (int ct = 0; ct < 4; ++ct) {
            acc[0][ct] = WMMA_BF16(A0.v, Bt[ct].v, acc[0][ct]);
            acc[1][ct] = WMMA_BF16(A1.v, Bt[ct].v, acc[1][ct]);
        }
    }

    // bias (column-constant per lane) + stage to LDS as bf16
#pragma unroll
    for (int ct = 0; ct < 4; ++ct) {
        float bc = bias[hcol[ct]];
#pragma unroll
        for (int rt = 0; rt < 2; ++rt) {
#pragma unroll
            for (int e = 0; e < 8; ++e) {
                Es[32 * wr + 16 * rt + e + 8 * kh][64 * wc + 16 * ct + row0] =
                    f2bf(acc[rt][ct][e] + bc);
            }
        }
    }
    __syncthreads();

    if (z != 2) {
        // row-major out, coalesced b128 stores
        int rr = tid >> 2, seg = tid & 3;
        int n = nbase + rr;
        if (n < NN) {
            u16* dst = outp + ((size_t)(b * NP + n)) * HH + hb + seg * 64;
#pragma unroll
            for (int qq = 0; qq < 8; ++qq)
                *(int4*)(dst + qq * 8) = *(const int4*)&Es[rr][seg * 64 + qq * 8];
        }
    } else {
        // transposed out: lvT[b][h][m]
        int hl = tid;   // one h row per thread (256 rows)
        u16* dst = outp + ((size_t)(b * HH + hb + hl)) * NP + nbase;
#pragma unroll
        for (int j = 0; j < 64; ++j) {
            int m = nbase + j;
            if (m < NP) dst[j] = Es[j][hl];
        }
    }
}

// ---------------------------------------------------------------------------
// Stage 2: qp = lq + param, ql = lq + lloc (bf16); outputs alias Xq/Xk.
// ---------------------------------------------------------------------------
__global__ __launch_bounds__(256) void k_combine(
    const float* __restrict__ param, const u16* __restrict__ lq,
    const u16* __restrict__ lloc, u16* __restrict__ qp, u16* __restrict__ ql)
{
    size_t i4 = ((size_t)blockIdx.x * 256 + threadIdx.x) * 4; // < 16*200*2048
    int b   = (int)(i4 / ((size_t)NN * HH));
    int rem = (int)(i4 % ((size_t)NN * HH));
    int n = rem >> 11;
    int h = rem & (HH - 1);
    size_t so = ((size_t)(b * NP + n)) * HH + h;
    float4  p = *(const float4*)(param + i4);
    ushort4 a = *(const ushort4*)(lq + so);
    ushort4 c = *(const ushort4*)(lloc + so);
    ushort4 o1, o2;
    o1.x = f2bf(bf2f(a.x) + p.x); o1.y = f2bf(bf2f(a.y) + p.y);
    o1.z = f2bf(bf2f(a.z) + p.z); o1.w = f2bf(bf2f(a.w) + p.w);
    o2.x = f2bf(bf2f(a.x) + bf2f(c.x)); o2.y = f2bf(bf2f(a.y) + bf2f(c.y));
    o2.z = f2bf(bf2f(a.z) + bf2f(c.z)); o2.w = f2bf(bf2f(a.w) + bf2f(c.w));
    *(ushort4*)(qp + so) = o1;
    *(ushort4*)(ql + so) = o2;
}

// ---------------------------------------------------------------------------
// Stage 3: fused attention per (batch, 16-row tile).
//   scores = qp@lk^T + ql@kloc^T (same f32 accumulators),
//   masked softmax over m (m<200), attn bf16 in LDS, O = (attn@lv)/rowsum.
// 256 threads = 8 waves; wave w owns m-columns [32w, 32w+32) of a 256 m-pad.
// B operands stream directly from L2-resident ws (b128, contiguous k = 16*kh + s order).
// ---------------------------------------------------------------------------
__global__ __launch_bounds__(256) void k_attn(
    const u16* __restrict__ qp, const u16* __restrict__ ql,
    const u16* __restrict__ lkp, const u16* __restrict__ klp,
    const u16* __restrict__ lvT, float* __restrict__ out)
{
    const int bx  = blockIdx.x;        // 0..12 (n-tile)
    const int b   = blockIdx.y;        // batch
    const int tid = threadIdx.x;
    const int lane = tid & 31, w = tid >> 5;
    const int row0 = lane & 15, kh = lane >> 4;

    __shared__ float S[16][260];
    __shared__ float P[16][16];
    __shared__ float rmaxs[16];
    __shared__ float rinvs[16];
    __shared__ u16   A16[16][264];

    v8f s0 = {}, s1 = {};
    const size_t aoff = ((size_t)(b * NP) + 16 * bx + row0) * HH;
    const int    m0   = 32 * w + row0;
    const size_t bo0  = ((size_t)(b * NP) + m0) * HH;
    const size_t bo1  = bo0 + (size_t)16 * HH;

    for (int kb = 0; kb < HH; kb += 32) {
        BfOp aq, al, K0, K1, C0, C1;
        aq.q[0] = *(const int4*)(qp + aoff + kb + 16 * kh);
        aq.q[1] = *(const int4*)(qp + aoff + kb + 16 * kh + 8);
        al.q[0] = *(const int4*)(ql + aoff + kb + 16 * kh);
        al.q[1] = *(const int4*)(ql + aoff + kb + 16 * kh + 8);
        K0.q[0] = *(const int4*)(lkp + bo0 + kb + 16 * kh);
        K0.q[1] = *(const int4*)(lkp + bo0 + kb + 16 * kh + 8);
        K1.q[0] = *(const int4*)(lkp + bo1 + kb + 16 * kh);
        K1.q[1] = *(const int4*)(lkp + bo1 + kb + 16 * kh + 8);
        C0.q[0] = *(const int4*)(klp + bo0 + kb + 16 * kh);
        C0.q[1] = *(const int4*)(klp + bo0 + kb + 16 * kh + 8);
        C1.q[0] = *(const int4*)(klp + bo1 + kb + 16 * kh);
        C1.q[1] = *(const int4*)(klp + bo1 + kb + 16 * kh + 8);
        s0 = WMMA_BF16(aq.v, K0.v, s0);
        s0 = WMMA_BF16(al.v, C0.v, s0);
        s1 = WMMA_BF16(aq.v, K1.v, s1);
        s1 = WMMA_BF16(al.v, C1.v, s1);
    }

    const float scale = 0.022097086912079608f;  // 1/sqrt(2048)
#pragma unroll
    for (int e = 0; e < 8; ++e) {
        S[e + 8 * kh][32 * w +  0 + row0] = s0[e] * scale;
        S[e + 8 * kh][32 * w + 16 + row0] = s1[e] * scale;
    }
    __syncthreads();

    // masked row max
    int r = tid >> 4, sg = tid & 15;
    float mx = -3.0e38f;
#pragma unroll
    for (int j = 0; j < 16; ++j) { int c = sg * 16 + j; if (c < NN) mx = fmaxf(mx, S[r][c]); }
    P[r][sg] = mx;
    __syncthreads();
    if (tid < 16) {
        float m2 = -3.0e38f;
        for (int j = 0; j < 16; ++j) m2 = fmaxf(m2, P[tid][j]);
        rmaxs[tid] = m2;
    }
    __syncthreads();

    // exp + row sum; bf16 weights (zero for masked columns)
    float rm = rmaxs[r];
    float psum = 0.f;
#pragma unroll
    for (int j = 0; j < 16; ++j) {
        int c = sg * 16 + j;
        float vv = (c < NN) ? __expf(S[r][c] - rm) : 0.f;
        A16[r][c] = f2bf(vv);
        psum += vv;
    }
    P[r][sg] = psum;
    __syncthreads();
    if (tid < 16) {
        float s = 0.f;
        for (int j = 0; j < 16; ++j) s += P[tid][j];
        rinvs[tid] = 1.0f / s;
    }
    __syncthreads();

    float rinv[8];
#pragma unroll
    for (int e = 0; e < 8; ++e) rinv[e] = rinvs[e + 8 * kh];

    // O = attn @ lv  (A from LDS, B from lvT in L2)
    for (int hc = 0; hc < HH; hc += 256) {
        v8f o0 = {}, o1 = {};
        const int h0 = hc + 32 * w + row0;
        const size_t vo0 = ((size_t)(b * HH) + h0) * NP;
        const size_t vo1 = vo0 + (size_t)16 * NP;
#pragma unroll
        for (int ms = 0; ms < 256; ms += 32) {
            BfOp av, V0, V1;
            av.q[0] = *(const int4*)&A16[row0][ms + 16 * kh];
            av.q[1] = *(const int4*)&A16[row0][ms + 16 * kh + 8];
            V0.q[0] = *(const int4*)(lvT + vo0 + ms + 16 * kh);
            V0.q[1] = *(const int4*)(lvT + vo0 + ms + 16 * kh + 8);
            V1.q[0] = *(const int4*)(lvT + vo1 + ms + 16 * kh);
            V1.q[1] = *(const int4*)(lvT + vo1 + ms + 16 * kh + 8);
            o0 = WMMA_BF16(av.v, V0.v, o0);
            o1 = WMMA_BF16(av.v, V1.v, o1);
        }
#pragma unroll
        for (int e = 0; e < 8; ++e) {
            int nn = 16 * bx + e + 8 * kh;
            if (nn < NN) {
                float* dst = out + ((size_t)(b * NN + nn)) * HH + hc + 32 * w + row0;
                dst[0]  = o0[e] * rinv[e];
                dst[16] = o1[e] * rinv[e];
            }
        }
    }
}

// ---------------------------------------------------------------------------
extern "C" void kernel_launch(void* const* d_in, const int* in_sizes, int n_in,
                              void* d_out, int out_size, void* d_ws, size_t ws_size,
                              hipStream_t stream)
{
    const float* qin  = (const float*)d_in[0];
    const float* kin  = (const float*)d_in[1];
    const float* vin  = (const float*)d_in[2];
    const float* locv = (const float*)d_in[3];
    const float* Wq   = (const float*)d_in[4];
    const float* bq   = (const float*)d_in[5];
    const float* Wk   = (const float*)d_in[6];
    const float* bk   = (const float*)d_in[7];
    const float* Wv   = (const float*)d_in[8];
    const float* bv   = (const float*)d_in[9];
    const float* Wloc = (const float*)d_in[10];
    const float* bloc = (const float*)d_in[11];
    const float* Wlk  = (const float*)d_in[12];
    const float* blk  = (const float*)d_in[13];
    const float* prm  = (const float*)d_in[14];
    float* out = (float*)d_out;

    const size_t SZ  = (size_t)NB * NP * HH;  // 6,815,744 elems per activation
    const size_t WSZ = (size_t)DD * HH;       // 4,194,304 elems per weight
    u16* ws    = (u16*)d_ws;
    u16* lq    = ws;             // [b][NP][H]
    u16* lk_   = ws + 1 * SZ;    // [b][NP][H]
    u16* lvT   = ws + 2 * SZ;    // [b][H][NP]  (tail over-reads land in lloc)
    u16* lloc  = ws + 3 * SZ;    // [b][NP][H]
    u16* kloc  = ws + 4 * SZ;    // [b][NP][H]  (tail over-reads land in Xq)
    u16* Xq    = ws + 5 * SZ;    // bf16 query; later reused as qp
    u16* Xk    = ws + 6 * SZ;    // bf16 key;   later reused as ql
    u16* Xv    = ws + 7 * SZ;    // bf16 value
    u16* Xl    = ws + 8 * SZ;    // bf16 loc
    u16* WTq   = ws + 9 * SZ;
    u16* WTk   = WTq + 1 * WSZ;
    u16* WTv   = WTq + 2 * WSZ;
    u16* WTloc = WTq + 3 * WSZ;
    u16* WTlk  = WTq + 4 * WSZ;
    u16* qp    = Xq;             // alias: Xq dead after k_proj
    u16* ql    = Xk;             // alias: Xk dead after k_proj

    kc_x<<<dim3(6656, 1, 4), 256, 0, stream>>>(qin, kin, vin, locv, Xq, Xk, Xv, Xl);

    kc_w<<<dim3(32, 32, 5), 256, 0, stream>>>(Wq, Wk, Wv, Wloc, Wlk,
                                              WTq, WTk, WTv, WTloc, WTlk);

    k_proj<<<dim3(8, 64, 5), 256, 0, stream>>>(
        Xq, Xk, Xv, Xl, WTq, WTk, WTv, WTloc, WTlk,
        bq, bk, bv, bloc, blk,
        lq, lk_, lvT, lloc, kloc);

    k_combine<<<6400, 256, 0, stream>>>(prm, lq, lloc, qp, ql);

    k_attn<<<dim3(13, NB), 256, 0, stream>>>(qp, ql, lk_, kloc, lvT, out);
}
